// GLifNet_23132693856378
// MI455X (gfx1250) — matrix operand order repaired and checked by reading, not compiled, not run on hardware
//
#include <hip/hip_runtime.h>
#include <math.h>

typedef __attribute__((ext_vector_type(16))) _Float16 v16h;
typedef __attribute__((ext_vector_type(8)))  float    v8f;

// ---- problem constants (GLIF3 reference) ----
#define N_NEU      2048
#define BATCH      64
#define TSTEPS     100
#define NUM_IN     784
#define NUM_IN_NEU 1024
#define NUM_OUT    10
#define NUM_OUT_NEU 128
#define XPAD       800                  // NUM_IN padded to mult of 32/8
#define KTOT       2944                 // 2048 (rec) + 800 (input) + 96 zero = 23*128
#define KT_TILES   (KTOT / 32)          // 92 k-tiles of 32
#define CHUNK      128                  // K staged per LDS pass (4 k-tiles)
#define NCHUNK     (KTOT / CHUNK)       // 23
#define NTILE      64                   // neuron columns per workgroup
#define APITCH     (CHUNK + 8)          // 136 halves: pad -> conflict-free ds_load_b128
#define CPITCH     (NTILE + 1)          // 65 floats

#define C_V_TH    (-45.0f)
#define C_V_RESET (-60.0f)
#define C_CM_INV  (0.5f)                // 1/C_M
#define C_TAU_INV (0.05f)               // 1/TAU
#define C_ASC_AMP (-0.2f)
#define C_D_ASC   (0.98757795f)         // exp(-1/80)
#define C_BASE_I  (0.12f)
#define C_TAU_REF (2.0f)
#define C_E       (2.7182818284590452f)

// ================= one-time prep kernels =================

__global__ void fill_maps(int* invmap, int* outmap) {
    int i = blockIdx.x * 256 + threadIdx.x;
    if (i < N_NEU) { invmap[i] = -1; outmap[i] = -1; }
}

__global__ void scatter_maps(const int* __restrict__ inIdx, const int* __restrict__ outIdx,
                             int* invmap, int* outmap) {
    int i = blockIdx.x * 256 + threadIdx.x;
    if (i < NUM_IN_NEU) invmap[inIdx[i]] = i;
    if (i < NUM_OUT_NEU) outmap[outIdx[i]] = i;
}

// Pack B' = [W_rec^T ; scattered W_in ; 0] into per-(n_tile,k_tile) WMMA B fragments.
// flat = ((nt*KT_TILES + kt)*32 + lane)*16 + h ; lane runtime-loads 16 contiguous halves.
// B layout (32x16 f16, K x N): col N = lane%16; lanes 0-15 hold K 0..7,16..23 (h<8: K=h, else K=8+h),
// lanes 16-31 hold K 8..15,24..31 (h<8: K=8+h, else K=16+h).
__global__ void pack_w(const float* __restrict__ Wrec, const float* __restrict__ Win,
                       const int* __restrict__ invmap, _Float16* __restrict__ Wp) {
    long long flat = (long long)blockIdx.x * 256 + threadIdx.x;
    if (flat >= (long long)N_NEU * KTOT) return;
    int h    = (int)(flat & 15);
    int lane = (int)((flat >> 4) & 31);
    int tcmb = (int)(flat >> 9);
    int kt = tcmb % KT_TILES;
    int nt = tcmb / KT_TILES;
    int n  = nt * 16 + (lane & 15);
    int k  = kt * 32 + (h & 7) + ((lane < 16) ? 0 : 8) + ((h >> 3) << 4);
    float val = 0.0f;
    if (k < N_NEU) {
        val = Wrec[(long long)n * N_NEU + k];                 // B[k][n] = W_rec[n][k]
    } else if (k < N_NEU + NUM_IN) {
        int j = invmap[n];                                    // input-neuron drive folded into K
        if (j >= 0) val = Win[(long long)j * NUM_IN + (k - N_NEU)];
    }
    Wp[flat] = (_Float16)val;
}

__global__ void pack_x(const float* __restrict__ x, _Float16* __restrict__ xp) {
    long long i = (long long)blockIdx.x * 256 + threadIdx.x;
    if (i >= (long long)TSTEPS * BATCH * XPAD) return;
    int col = (int)(i % XPAD);
    long long tb = i / XPAD;
    float val = (col < NUM_IN) ? x[tb * NUM_IN + col] : 0.0f;
    xp[i] = (_Float16)val;
}

__global__ void init_state(const float* __restrict__ tau,
                           float* v, float* psc, float* rise, float* asc, float* ref,
                           float* dsyn, float* etau, float* rate, _Float16* spk0) {
    int i = blockIdx.x * 256 + threadIdx.x;            // 0 .. BATCH*N_NEU-1
    v[i]    = C_V_RESET;
    psc[i]  = 0.0f;  rise[i] = 0.0f;  asc[i] = 0.0f;  ref[i] = 0.0f;
    spk0[i] = (_Float16)0.0f;
    if (i < BATCH * NUM_OUT_NEU) rate[i] = 0.0f;
    if (i < N_NEU) {
        float ts = tau[i];
        dsyn[i] = expf(-1.0f / ts);
        etau[i] = C_E / ts;
    }
}

// ================= fused per-step kernel =================
// grid = 32 blocks (64 neuron cols each) x 256 threads (8 waves).
// Wave w: M-tile mi = w&3, N-tiles {njb, njb+1}, njb = (w>>2)*2.
__global__ __launch_bounds__(256) void glif_step(
        const _Float16* __restrict__ Wp, const _Float16* __restrict__ xp,
        const _Float16* __restrict__ spk_rd, _Float16* __restrict__ spk_wr,
        float* __restrict__ v, float* __restrict__ psc, float* __restrict__ rise,
        float* __restrict__ asc, float* __restrict__ ref,
        const float* __restrict__ dsyn, const float* __restrict__ etau,
        const int* __restrict__ outmap, float* __restrict__ rate_acc, int t) {
    __shared__ __align__(16) _Float16 ldsA[BATCH * APITCH];   // 17.0 KB staged A' slab
    __shared__ float ldsC[BATCH * CPITCH];                    // 16.25 KB f32 C tile

    const int tid  = threadIdx.x;
    const int lane = tid & 31;
    const int w    = tid >> 5;
    const int n0   = blockIdx.x * NTILE;
    const int mi   = w & 3;
    const int njb  = (w >> 2) * 2;
    const int rowl = lane & 15;
    const int hoff = (lane < 16) ? 0 : 8;   // A frag: lanes 0-15 K{0..7,16..23}, 16-31 K{8..15,24..31}

    v8f c0 = {}; v8f c1 = {};

    const uint4* spk4 = (const uint4*)spk_rd;
    const uint4* xp4  = (const uint4*)(xp + (long long)t * BATCH * XPAD);
    const uint4* wp4  = (const uint4*)Wp;
    uint4* ldsA4 = (uint4*)ldsA;

    for (int kc = 0; kc < NCHUNK; ++kc) {
        // ---- stage A' chunk (64 x 128 f16): spikes | x_t | zero-pad ----
        #pragma unroll
        for (int q = 0; q < 4; ++q) {
            int s   = tid + q * 256;          // 1024 uint4 slots
            int row = s >> 4;                 // batch row
            int c8  = (s & 15) * 8;           // half offset in chunk
            int kg  = kc * CHUNK + c8;        // global K
            uint4 val;
            if (kg < N_NEU) {
                val = spk4[(row * N_NEU + kg) >> 3];
            } else if (kg < N_NEU + XPAD) {
                val = xp4[(row * XPAD + (kg - N_NEU)) >> 3];
            } else {
                val = make_uint4(0u, 0u, 0u, 0u);
            }
            ldsA4[(row * APITCH + c8) >> 3] = val;
        }
        __syncthreads();
        // ---- 4 k-tiles of WMMA ----
        #pragma unroll
        for (int ki = 0; ki < 4; ++ki) {
            const int kk = ki * 32;
            union { v16h h; uint4 u[2]; } a, b0, b1;
            int abase = ((mi * 16 + rowl) * APITCH + kk + hoff) >> 3;
            a.u[0] = ldsA4[abase];           // K kk+hoff .. +7
            a.u[1] = ldsA4[abase + 2];       // K kk+16+hoff .. +7
            int ktg  = kc * 4 + ki;
            int t0i  = ((n0 >> 4) + njb) * KT_TILES + ktg;
            int bi0  = (t0i * 32 + lane) * 2;
            int bi1  = ((t0i + KT_TILES) * 32 + lane) * 2;
            b0.u[0] = wp4[bi0];     b0.u[1] = wp4[bi0 + 1];
            b1.u[0] = wp4[bi1];     b1.u[1] = wp4[bi1 + 1];
            c0 = __builtin_amdgcn_wmma_f32_16x16x32_f16(false, a.h, false, b0.h,
                                                        (short)0, c0, false, false);
            c1 = __builtin_amdgcn_wmma_f32_16x16x32_f16(false, a.h, false, b1.h,
                                                        (short)0, c1, false, false);
        }
        __syncthreads();
    }

    // ---- spill C tiles to LDS (ISA C layout: reg r -> M = mi*16 + r (+8 for hi lanes)) ----
    {
        int mrow  = mi * 16 + ((lane < 16) ? 0 : 8);
        int ncol0 = njb * 16 + rowl;
        #pragma unroll
        for (int r = 0; r < 8; ++r) {
            ldsC[(mrow + r) * CPITCH + ncol0]      = c0[r];
            ldsC[(mrow + r) * CPITCH + ncol0 + 16] = c1[r];
        }
    }
    __syncthreads();

    // ---- GLIF3 elementwise update for this block's (b, n) range ----
    #pragma unroll
    for (int q = 0; q < 16; ++q) {
        int pos = tid + q * 256;              // 0..4095
        int b   = pos >> 6;
        int nl  = pos & 63;
        int n   = n0 + nl;
        long long gi = (long long)b * N_NEU + n;
        float Iin = ldsC[b * CPITCH + nl];    // recurrent + input drive (folded K)
        float ds  = dsyn[n];
        float et  = etau[n];
        float vv  = v[gi],  p = psc[gi], rs = rise[gi], ac = asc[gi], rf = ref[gi];
        float spk_old = (float)spk_rd[gi];
        float p_n = ds * (p + rs);                          // alpha-PSC, DT = 1
        float r_n = ds * rs + et * Iin;
        float a_n = C_D_ASC * ac + C_ASC_AMP * spk_old;     // after-spike current
        float active = (rf <= 0.0f) ? 1.0f : 0.0f;          // refractory gate
        float dv  = (p_n + a_n + C_BASE_I) * C_CM_INV - (vv - C_V_RESET) * C_TAU_INV;
        float v_n = vv + active * dv;
        float s_n = ((v_n - C_V_TH) >= 0.0f) ? active : 0.0f;
        v_n -= s_n * (C_V_TH - C_V_RESET);                  // soft reset
        float rf_n = (s_n > 0.0f) ? C_TAU_REF : fmaxf(rf - 1.0f, 0.0f);
        v[gi] = v_n; psc[gi] = p_n; rise[gi] = r_n; asc[gi] = a_n; ref[gi] = rf_n;
        spk_wr[gi] = (_Float16)s_n;
        int om = outmap[n];
        if (om >= 0) rate_acc[b * NUM_OUT_NEU + om] += s_n; // unique owner -> no atomics
    }
}

// ================= readout =================
__global__ void out_proj(const float* __restrict__ rate, const float* __restrict__ Wout,
                         float* __restrict__ out) {
    int i = blockIdx.x * 256 + threadIdx.x;
    if (i >= BATCH * NUM_OUT) return;
    int b = i / NUM_OUT, o = i % NUM_OUT;
    float acc = 0.0f;
    for (int p = 0; p < NUM_OUT_NEU; ++p)
        acc += rate[b * NUM_OUT_NEU + p] * Wout[o * NUM_OUT_NEU + p];
    out[i] = acc * (1.0f / TSTEPS);
}

// ================= host launcher =================
extern "C" void kernel_launch(void* const* d_in, const int* in_sizes, int n_in,
                              void* d_out, int out_size, void* d_ws, size_t ws_size,
                              hipStream_t stream) {
    const float* x      = (const float*)d_in[0];
    const float* Win    = (const float*)d_in[1];
    const float* Wrec   = (const float*)d_in[2];
    const float* Wout   = (const float*)d_in[3];
    const float* tau    = (const float*)d_in[4];
    const int*   inIdx  = (const int*)d_in[5];
    const int*   outIdx = (const int*)d_in[6];
    float* out = (float*)d_out;

    // bump allocator over d_ws (256B aligned slices), total ~25.5 MB
    char* p = (char*)d_ws;
    auto alloc = [&](size_t bytes) -> void* {
        void* r = (void*)p;
        p += (bytes + 255) & ~((size_t)255);
        return r;
    };
    _Float16* Wp   = (_Float16*)alloc(sizeof(_Float16) * (size_t)N_NEU * KTOT);
    _Float16* xp   = (_Float16*)alloc(sizeof(_Float16) * (size_t)TSTEPS * BATCH * XPAD);
    float* v    = (float*)alloc(sizeof(float) * BATCH * N_NEU);
    float* psc  = (float*)alloc(sizeof(float) * BATCH * N_NEU);
    float* rise = (float*)alloc(sizeof(float) * BATCH * N_NEU);
    float* asc  = (float*)alloc(sizeof(float) * BATCH * N_NEU);
    float* ref  = (float*)alloc(sizeof(float) * BATCH * N_NEU);
    _Float16* spkA = (_Float16*)alloc(sizeof(_Float16) * BATCH * N_NEU);
    _Float16* spkB = (_Float16*)alloc(sizeof(_Float16) * BATCH * N_NEU);
    float* rate = (float*)alloc(sizeof(float) * BATCH * NUM_OUT_NEU);
    float* dsyn = (float*)alloc(sizeof(float) * N_NEU);
    float* etau = (float*)alloc(sizeof(float) * N_NEU);
    int* invmap = (int*)alloc(sizeof(int) * N_NEU);
    int* outmap = (int*)alloc(sizeof(int) * N_NEU);
    _Float16* spk[2] = { spkA, spkB };

    fill_maps<<<N_NEU / 256, 256, 0, stream>>>(invmap, outmap);
    scatter_maps<<<NUM_IN_NEU / 256, 256, 0, stream>>>(inIdx, outIdx, invmap, outmap);
    pack_w<<<(int)(((long long)N_NEU * KTOT) / 256), 256, 0, stream>>>(Wrec, Win, invmap, Wp);
    pack_x<<<(int)(((long long)TSTEPS * BATCH * XPAD) / 256), 256, 0, stream>>>(x, xp);
    init_state<<<(BATCH * N_NEU) / 256, 256, 0, stream>>>(tau, v, psc, rise, asc, ref,
                                                          dsyn, etau, rate, spkA);
    for (int t = 0; t < TSTEPS; ++t) {
        glif_step<<<N_NEU / NTILE, 256, 0, stream>>>(
            Wp, xp, spk[t & 1], spk[(t + 1) & 1],
            v, psc, rise, asc, ref, dsyn, etau, outmap, rate, t);
    }
    out_proj<<<(BATCH * NUM_OUT + 255) / 256, 256, 0, stream>>>(rate, Wout, out);
}
